// Com_AttFusion_39152921870870
// MI455X (gfx1250) — matrix operand order repaired and checked by reading, not compiled, not run on hardware
//
#include <hip/hip_runtime.h>
#include <math.h>
#include <stdint.h>

// CDNA5 (gfx1250) wave32 WMMA implementation of the fused score-select /
// attention / Mamba-2 SSD block. All GEMMs use v_wmma_f32_16x16x32_f16, and
// the SSD chunk kernels stage their LDS tiles with global_load_async_to_lds
// (ASYNCcnt) overlapped with the in-chunk cumsum scan.
// Note: the reference softmax is over a singleton axis => attn == 1 exactly,
// so ctx == v and the q/k projections are dead code (skipped, exact math).

typedef __attribute__((ext_vector_type(16))) _Float16 v16h;
typedef __attribute__((ext_vector_type(8)))  float    v8f;

namespace cfg {
constexpr int C = 64, H = 128, Wd = 256, HW = H * Wd;   // HW = 32768
constexpr int DIN = 128, NH = 2, DST = 64;
constexpr int CONVD = 256, DPROJ = 386;
constexpr int CHUNK = 256, NCH = HW / CHUNK;            // 128 chunks
constexpr float EPS = 1e-5f;
}

#define DEV __device__ __forceinline__

DEV int lane_id() { return (int)(threadIdx.x & 31); }

// ---- CDNA5 async global->LDS copy (tracked by ASYNCcnt) -------------------
// lds_off: byte offset in LDS (low 32 bits of a generic shared pointer;
// for the LDS aperture the low 32 bits ARE the LDS address).
DEV void async_copy_b128(uint32_t lds_off, const void* gptr) {
  asm volatile("global_load_async_to_lds_b128 %0, %1, off"
               :: "v"(lds_off), "v"((uint64_t)(uintptr_t)gptr)
               : "memory");
}
DEV void async_wait0() { asm volatile("s_wait_asynccnt 0x0" ::: "memory"); }
DEV uint32_t lds_off_of(const void* p) { return (uint32_t)(uintptr_t)p; }

// D = A(16x32) * B(32x16) + C.  A staged row-major (M x K, lda in halves).
// B given as weight rows: (N x K) row-major, ldb in halves (i.e. Y = A * W^T).
DEV v8f wmma_ab_nt(v8f acc, const _Float16* A, int lda, const _Float16* Bt, int ldb) {
  const int ln = lane_id(), r = ln & 15, hi = ln >> 4;
  v16h a, b;
  const _Float16* ap = A + r * lda + hi * 16;   // K: hi*16 + 0..15
  const _Float16* bp = Bt + r * ldb + hi * 16;  // b[j] = W[n=r][k=hi*16+j]
#pragma unroll
  for (int j = 0; j < 16; ++j) { a[j] = ap[j]; }
#pragma unroll
  for (int j = 0; j < 16; ++j) { b[j] = bp[j]; }
  return __builtin_amdgcn_wmma_f32_16x16x32_f16(false, a, false, b, (short)0, acc, false, false);
}

// D = A(16x32) * B(32x16) + C.  B staged row-major (K x N, ldb in halves).
DEV v8f wmma_ab_nn(v8f acc, const _Float16* A, int lda, const _Float16* B, int ldb) {
  const int ln = lane_id(), r = ln & 15, hi = ln >> 4;
  v16h a, b;
  const _Float16* ap = A + r * lda + hi * 16;
#pragma unroll
  for (int j = 0; j < 16; ++j) { a[j] = ap[j]; }
#pragma unroll
  for (int j = 0; j < 16; ++j) { b[j] = B[(hi * 16 + j) * ldb + r]; }
  return __builtin_amdgcn_wmma_f32_16x16x32_f16(false, a, false, b, (short)0, acc, false, false);
}

// D = A^T * B + C, with A staged as (K x M) row-major (lda in halves),
// B staged row-major (K x N).
DEV v8f wmma_at_nn(v8f acc, const _Float16* At, int lda, const _Float16* B, int ldb) {
  const int ln = lane_id(), r = ln & 15, hi = ln >> 4;
  v16h a, b;
#pragma unroll
  for (int j = 0; j < 16; ++j) {
    a[j]     = At[(hi * 16 + j) * lda + r];
  }
#pragma unroll
  for (int j = 0; j < 16; ++j) { b[j] = B[(hi * 16 + j) * ldb + r]; }
  return __builtin_amdgcn_wmma_f32_16x16x32_f16(false, a, false, b, (short)0, acc, false, false);
}

DEV float siluf(float x) { return x / (1.f + __expf(-x)); }

// ---------------------------------------------------------------------------
// K1: per-pixel scoring head: score = w2 . relu(W1 x + b1) + b2   (WMMA GEMM)
// grid (HW/64, NREC), block 256
__global__ void k_scores(const float* __restrict__ xx, const float* __restrict__ w1,
                         const float* __restrict__ b1, const float* __restrict__ w2,
                         const float* __restrict__ b2, float* __restrict__ scores) {
  using namespace cfg;
  __shared__ _Float16 Xs[64 * 64];   // [pixel][chan]
  __shared__ _Float16 Ws[64 * 64];   // W1 rows (N x K)
  __shared__ float sAcc[64];
  const int tid = threadIdx.x, wid = tid >> 5, r = tid & 15, hi = (tid >> 4) & 1;
  const int n = blockIdx.y;
  const size_t p0 = (size_t)blockIdx.x * 64;
  for (int idx = tid; idx < 4096; idx += 256) {
    int c = idx >> 6, p = idx & 63;
    Xs[p * 64 + c] = (_Float16)xx[((size_t)n * 64 + c) * HW + p0 + p];
    Ws[idx] = (_Float16)w1[idx];
  }
  if (tid < 64) sAcc[tid] = 0.f;
  __syncthreads();
  for (int t = wid; t < 16; t += 8) {
    int tm = t >> 2, tn = t & 3;
    v8f acc = {};
#pragma unroll
    for (int kk = 0; kk < 64; kk += 32)
      acc = wmma_ab_nt(acc, &Xs[tm * 16 * 64 + kk], 64, &Ws[tn * 16 * 64 + kk], 64);
    int d = tn * 16 + r;
    float w2d = w2[d], b1d = b1[d];
#pragma unroll
    for (int e = 0; e < 8; ++e) {
      float s = fmaxf(acc[e] + b1d, 0.f);
      atomicAdd(&sAcc[tm * 16 + e + 8 * hi], w2d * s);
    }
  }
  __syncthreads();
  if (tid < 64) scores[(size_t)n * HW + p0 + tid] = sAcc[tid] + b2[0];
}

// ---------------------------------------------------------------------------
// K2: per-group argmax over records, gather fused frame -> seqT[b][c][p]
// grid (HW/256, B), block 256
__global__ void k_select(const float* __restrict__ xx, const int* __restrict__ rl,
                         const float* __restrict__ scores, float* __restrict__ seqT) {
  using namespace cfg;
  const int p = blockIdx.x * 256 + threadIdx.x;
  const int b = blockIdx.y;
  int start = 0;
  for (int i = 0; i < b; ++i) start += rl[i];
  const int n = rl[b];
  int best = 0;
  float bs = scores[(size_t)start * HW + p];
  for (int i = 1; i < n; ++i) {
    float s = scores[(size_t)(start + i) * HW + p];
    if (s > bs) { bs = s; best = i; }   // first-max semantics like jnp.argmax
  }
  const float* src = xx + (size_t)(start + best) * 64 * HW + p;
  float* dst = seqT + (size_t)b * 64 * HW + p;
#pragma unroll 4
  for (int c = 0; c < 64; ++c) dst[(size_t)c * HW] = src[(size_t)c * HW];
}

// ---------------------------------------------------------------------------
// K3: ctx = seq @ v_w^T + v_b (attn==1), then LayerNorm -> u (f16) ; keep ctx f32
// grid (HW/64, B), block 256
__global__ void k_ctx_ln(const float* __restrict__ seqT, const float* __restrict__ vw,
                         const float* __restrict__ vb, const float* __restrict__ lnw,
                         const float* __restrict__ lnb, float* __restrict__ ctx,
                         _Float16* __restrict__ uh) {
  using namespace cfg;
  __shared__ _Float16 Xs[4096], Ws[4096];
  __shared__ float Cs[4096];
  const int tid = threadIdx.x, wid = tid >> 5, r = tid & 15, hi = (tid >> 4) & 1;
  const int b = blockIdx.y;
  const size_t p0 = (size_t)blockIdx.x * 64;
  for (int idx = tid; idx < 4096; idx += 256) {
    int c = idx >> 6, p = idx & 63;
    Xs[p * 64 + c] = (_Float16)seqT[((size_t)b * 64 + c) * HW + p0 + p];
    Ws[idx] = (_Float16)vw[idx];
  }
  __syncthreads();
  for (int t = wid; t < 16; t += 8) {
    int tm = t >> 2, tn = t & 3;
    v8f acc = {};
#pragma unroll
    for (int kk = 0; kk < 64; kk += 32)
      acc = wmma_ab_nt(acc, &Xs[tm * 16 * 64 + kk], 64, &Ws[tn * 16 * 64 + kk], 64);
    int d = tn * 16 + r;
    float bd = vb[d];
#pragma unroll
    for (int e = 0; e < 8; ++e) Cs[(tm * 16 + e + 8 * hi) * 64 + d] = acc[e] + bd;
  }
  __syncthreads();
  if (tid < 64) {
    const float* row = &Cs[tid * 64];
    float mu = 0.f, m2 = 0.f;
    for (int d = 0; d < 64; ++d) { mu += row[d]; m2 += row[d] * row[d]; }
    mu *= (1.f / 64.f); m2 *= (1.f / 64.f);
    float inv = rsqrtf(m2 - mu * mu + EPS);
    size_t pix = (size_t)b * HW + p0 + tid;
    for (int d = 0; d < 64; ++d) {
      float x = row[d];
      ctx[pix * 64 + d] = x;
      uh[pix * 64 + d] = (_Float16)((x - mu) * inv * lnw[d] + lnb[d]);
    }
  }
}

// ---------------------------------------------------------------------------
// K4: zxbcdt = u @ in_proj_w^T + b ; route: silu(z), xBC (channel-major for
// conv), dt -> softplus(dt+bias), dA = dt * (-exp(A_log)).
// grid (HW/64, B), block 256
__global__ void k_inproj(const _Float16* __restrict__ uh, const float* __restrict__ wip,
                         const float* __restrict__ bip, const float* __restrict__ dtb,
                         const float* __restrict__ alog, float* __restrict__ zs,
                         float* __restrict__ xBCt, float* __restrict__ dtvg,
                         float* __restrict__ dAg) {
  using namespace cfg;
  __shared__ _Float16 Xs[64 * 64];
  __shared__ _Float16 Ws[400 * 64];          // 386 rows, zero-padded to 400
  const int tid = threadIdx.x, wid = tid >> 5, r = tid & 15, hi = (tid >> 4) & 1;
  const int b = blockIdx.y;
  const size_t p0 = (size_t)blockIdx.x * 64;
  for (int idx = tid; idx < 4096; idx += 256) {
    int m = idx >> 6, c = idx & 63;
    Xs[idx] = uh[((size_t)b * HW + p0 + m) * 64 + c];
  }
  for (int idx = tid; idx < 400 * 64; idx += 256) {
    int n = idx >> 6;
    Ws[idx] = (n < DPROJ) ? (_Float16)wip[idx] : (_Float16)0.f;
  }
  __syncthreads();
  for (int t = wid; t < 100; t += 8) {       // 4 M-tiles x 25 N-tiles
    int tm = t / 25, tn = t % 25;
    v8f acc = {};
#pragma unroll
    for (int kk = 0; kk < 64; kk += 32)
      acc = wmma_ab_nt(acc, &Xs[tm * 16 * 64 + kk], 64, &Ws[tn * 16 * 64 + kk], 64);
    int n = tn * 16 + r;
    float bn = (n < DPROJ) ? bip[n] : 0.f;
#pragma unroll
    for (int e = 0; e < 8; ++e) {            // per-lane guards only around stores
      int m = tm * 16 + e + 8 * hi;
      size_t pix = (size_t)b * HW + p0 + m;
      if (n < DPROJ) {
        float v = acc[e] + bn;
        if (n < DIN) {
          zs[pix * DIN + n] = siluf(v);                       // gate, pre-SiLU'd
        } else if (n < DIN + CONVD) {
          int ch = n - DIN;
          xBCt[((size_t)b * CONVD + ch) * HW + p0 + m] = v;   // channel-major
        } else {
          int hh = n - (DIN + CONVD);
          float x = v + dtb[hh];
          float dtv = (x > 20.f) ? x : log1pf(__expf(x));     // softplus
          dtvg[pix * NH + hh] = dtv;
          dAg[pix * NH + hh] = dtv * (-__expf(alog[hh]));
        }
      }
    }
  }
}

// ---------------------------------------------------------------------------
// K5: causal depthwise conv (K=4) + SiLU; split into xh(f32), xdt=xh*dt (f16),
// B (f16), C (f16).   grid (HW/256, CONVD, B), block 256
__global__ void k_conv(const float* __restrict__ xBCt, const float* __restrict__ cw,
                       const float* __restrict__ cb, const float* __restrict__ dtvg,
                       float* __restrict__ xhg, _Float16* __restrict__ xdtg,
                       _Float16* __restrict__ Bhg, _Float16* __restrict__ Chg) {
  using namespace cfg;
  const int p = blockIdx.x * 256 + threadIdx.x;
  const int ch = blockIdx.y, b = blockIdx.z;
  const float* xrow = xBCt + ((size_t)b * CONVD + ch) * HW;
  float acc = cb[ch];
#pragma unroll
  for (int k = 0; k < 4; ++k) {
    int q = p - 3 + k;
    if (q >= 0) acc += cw[ch * 4 + k] * xrow[q];
  }
  float v = siluf(acc);
  size_t pix = (size_t)b * HW + p;
  if (ch < DIN) {
    int hh = ch >> 6;
    xhg[pix * DIN + ch] = v;
    xdtg[pix * DIN + ch] = (_Float16)(v * dtvg[pix * NH + hh]);
  } else if (ch < DIN + DST) {
    Bhg[pix * DST + (ch - DIN)] = (_Float16)v;
  } else {
    Chg[pix * DST + (ch - DIN - DST)] = (_Float16)v;
  }
}

// ---------------------------------------------------------------------------
// K6a: per-(b,chunk,head): cumsum(dA), CBL = (C B^T) * exp-mask, Y_diag =
// CBL @ xdt + D*xh, chunk states = B^T @ (decay*xdt).  All GEMMs via WMMA.
// B/C/xdt tiles staged with async global->LDS copies, overlapped with the
// cumsum scan and the CBL zero-fill, fenced by s_wait_asynccnt + barrier.
// grid (NCH, NH, B), block 256, dynamic LDS ~230KB (one WG per WGP).
__global__ void k_chunk(const float* __restrict__ dAg, const _Float16* __restrict__ Bhg,
                        const _Float16* __restrict__ Chg, const _Float16* __restrict__ xdtg,
                        const float* __restrict__ xhg, const float* __restrict__ ssdD,
                        float* __restrict__ Acumg, float* __restrict__ Aendg,
                        float* __restrict__ states, float* __restrict__ yg) {
  using namespace cfg;
  extern __shared__ __align__(16) char smem[];
  _Float16* Bs  = (_Float16*)smem;            // 256x64
  _Float16* Cs  = Bs + 16384;                 // 256x64
  _Float16* Xs  = Cs + 16384;                 // 256x64 (xdt, this head)
  _Float16* CBL = Xs + 16384;                 // 256x256 masked C.B^T (f16)
  float*    Ac  = (float*)(CBL + 65536);      // 256 cumsum
  const int tid = threadIdx.x, wid = tid >> 5, r = tid & 15, hi = (tid >> 4) & 1;
  const int chunk = blockIdx.x, h = blockIdx.y, b = blockIdx.z;
  const size_t pbase = (size_t)b * HW + (size_t)chunk * CHUNK;

  // ---- kick off async HBM->LDS staging (each tile is 32KB = 2048 x 16B) ----
  {
    const char* bsrc = (const char*)(Bhg + pbase * DST);
    const char* csrc = (const char*)(Chg + pbase * DST);
    const uint32_t bs_off = lds_off_of(Bs), cs_off = lds_off_of(Cs);
    for (int i = tid; i < 2048; i += 256) {
      async_copy_b128(bs_off + i * 16, bsrc + i * 16);
      async_copy_b128(cs_off + i * 16, csrc + i * 16);
    }
    const uint32_t xs_off = lds_off_of(Xs);
    for (int i = tid; i < 2048; i += 256) {          // row l, 16B chunk c
      int l = i >> 3, c = i & 7;
      const char* src = (const char*)(xdtg + (pbase + l) * DIN + h * 64) + c * 16;
      async_copy_b128(xs_off + i * 16, src);
    }
  }

  // ---- overlapped: inclusive cumsum of dA over the chunk (Hillis-Steele) ---
  Ac[tid] = dAg[(pbase + tid) * NH + h];
  __syncthreads();
  for (int off = 1; off < 256; off <<= 1) {
    float t = (tid >= off) ? Ac[tid - off] : 0.f;
    __syncthreads();
    Ac[tid] += t;
    __syncthreads();
  }
  Acumg[(pbase + tid) * NH + h] = Ac[tid];
  if (tid == 255) Aendg[((size_t)b * NCH + chunk) * NH + h] = Ac[255];
  const float aend = Ac[255];

  // ---- overlapped: zero CBL (upper triangle must read as 0 in the GEMM) ----
  for (int idx = tid; idx < 65536; idx += 256) CBL[idx] = (_Float16)0.f;

  async_wait0();          // per-wave ASYNCcnt drain, then make visible to WG
  __syncthreads();

  // CBL[l][s] = (Cm[l].Bm[s]) * exp(Ac[l]-Ac[s]) on lower triangle
  for (int t = wid; t < 256; t += 8) {
    int li = t >> 4, si = t & 15;
    if (si > li) continue;                    // wave-uniform branch
    v8f acc = {};
#pragma unroll
    for (int kk = 0; kk < 64; kk += 32)
      acc = wmma_ab_nt(acc, &Cs[li * 16 * 64 + kk], 64, &Bs[si * 16 * 64 + kk], 64);
    int s = si * 16 + r;
#pragma unroll
    for (int e = 0; e < 8; ++e) {
      int l = li * 16 + e + 8 * hi;
      float v = (s <= l) ? acc[e] * __expf(Ac[l] - Ac[s]) : 0.f;
      CBL[l * 256 + s] = (_Float16)v;
    }
  }
  __syncthreads();

  // Y_diag = CBL @ xdt (+ D*xh)
  const float Dv = ssdD[h];
  for (int t = wid; t < 64; t += 8) {
    int li = t >> 2, pi = t & 3;
    v8f acc = {};
#pragma unroll
    for (int kk = 0; kk < 256; kk += 32)
      acc = wmma_ab_nn(acc, &CBL[li * 16 * 256 + kk], 256, &Xs[kk * 64 + pi * 16], 64);
    int pd = pi * 16 + r;
#pragma unroll
    for (int e = 0; e < 8; ++e) {
      int l = li * 16 + e + 8 * hi;
      size_t pix = pbase + l;
      float xv = xhg[pix * DIN + h * 64 + pd];
      yg[pix * DIN + h * 64 + pd] = acc[e] + Dv * xv;
    }
  }
  __syncthreads();

  // xdd[l][p] = exp(aend - Ac[l]) * xdt[l][p]  (reuse CBL space)
  _Float16* Xd = CBL;
  for (int idx = tid; idx < 16384; idx += 256) {
    int l = idx >> 6;
    Xd[idx] = (_Float16)(__expf(aend - Ac[l]) * (float)Xs[idx]);
  }
  __syncthreads();

  // chunk state S[n][p] = sum_l Bm[l][n] * xdd[l][p]
  float* Sout = states + (((size_t)b * NCH + chunk) * NH + h) * 4096;
  for (int t = wid; t < 16; t += 8) {
    int ni = t >> 2, pi = t & 3;
    v8f acc = {};
#pragma unroll
    for (int kk = 0; kk < 256; kk += 32)
      acc = wmma_at_nn(acc, &Bs[kk * 64 + ni * 16], 64, &Xd[kk * 64 + pi * 16], 64);
    int pd = pi * 16 + r;
#pragma unroll
    for (int e = 0; e < 8; ++e) {
      int n = ni * 16 + e + 8 * hi;
      Sout[n * 64 + pd] = acc[e];
    }
  }
}

// ---------------------------------------------------------------------------
// K6b: sequential inter-chunk recurrence, in place:
//   S_in[z] = prefix of states;  S <- exp(Aend_z)*S + states_z
// grid (B*NH*4096/256), block 256
__global__ void k_scan(float* __restrict__ states, const float* __restrict__ Aendg) {
  using namespace cfg;
  const int idx = blockIdx.x * 256 + threadIdx.x;
  const int e = idx & 4095, bh = idx >> 12;
  const int h = bh % NH, b = bh / NH;
  float S = 0.f;
  for (int z = 0; z < NCH; ++z) {
    size_t o = (((size_t)b * NCH + z) * NH + h) * 4096 + e;
    if (z + 1 < NCH)
      __builtin_prefetch(&states[o + (size_t)NH * 4096], 0, 1);  // global_prefetch_b8
    float old = states[o];
    states[o] = S;                                  // state entering chunk z
    S = __expf(Aendg[((size_t)b * NCH + z) * NH + h]) * S + old;
  }
}

// ---------------------------------------------------------------------------
// K6c: Y_off[l][p] = exp(Ac[l]) * sum_n Cm[l][n] * S_in[n][p], accumulate to y
// grid (NCH, NH, B), block 256
__global__ void k_yoff(const _Float16* __restrict__ Chg, const float* __restrict__ states,
                       const float* __restrict__ Acumg, float* __restrict__ yg) {
  using namespace cfg;
  __shared__ __align__(16) _Float16 Cs[256 * 64];
  __shared__ _Float16 Ss[64 * 64];    // [n][p]
  __shared__ float Ac[256];
  const int tid = threadIdx.x, wid = tid >> 5, r = tid & 15, hi = (tid >> 4) & 1;
  const int chunk = blockIdx.x, h = blockIdx.y, b = blockIdx.z;
  const size_t pbase = (size_t)b * HW + (size_t)chunk * CHUNK;
  const float* Sin = states + (((size_t)b * NCH + chunk) * NH + h) * 4096;
  {  // async-stage the contiguous 32KB C tile, overlap with Ss conversion
    const char* csrc = (const char*)(Chg + pbase * DST);
    const uint32_t cs_off = lds_off_of(Cs);
    for (int i = tid; i < 2048; i += 256)
      async_copy_b128(cs_off + i * 16, csrc + i * 16);
  }
  for (int idx = tid; idx < 4096; idx += 256) Ss[idx] = (_Float16)Sin[idx];
  Ac[tid] = Acumg[(pbase + tid) * NH + h];
  async_wait0();
  __syncthreads();
  for (int t = wid; t < 64; t += 8) {
    int li = t >> 2, pi = t & 3;
    v8f acc = {};
#pragma unroll
    for (int kk = 0; kk < 64; kk += 32)
      acc = wmma_ab_nn(acc, &Cs[li * 16 * 64 + kk], 64, &Ss[kk * 64 + pi * 16], 64);
    int pd = pi * 16 + r;
#pragma unroll
    for (int e = 0; e < 8; ++e) {
      int l = li * 16 + e + 8 * hi;
      size_t pix = pbase + l;
      yg[pix * DIN + h * 64 + pd] += __expf(Ac[l]) * acc[e];
    }
  }
}

// ---------------------------------------------------------------------------
// K7: yz = y*silu(z); RMS-norm * rms_w; h = ctx + yn @ out_proj^T; write
// transposed to (B,C,H,W).  grid (HW/64, B), block 256, dyn LDS 48KB
__global__ void k_out(const float* __restrict__ yg, const float* __restrict__ zs,
                      const float* __restrict__ rmsw, const float* __restrict__ wop,
                      const float* __restrict__ ctx, float* __restrict__ out) {
  using namespace cfg;
  extern __shared__ __align__(16) char smem[];
  float* Yz = (float*)smem;                        // 64 x 128 f32 (reused as f16)
  _Float16* Ws = (_Float16*)(smem + 64 * 128 * 4); // out_proj rows (64 x 128)
  const int tid = threadIdx.x, wid = tid >> 5, r = tid & 15, hi = (tid >> 4) & 1;
  const int b = blockIdx.y;
  const size_t p0 = (size_t)blockIdx.x * 64;
  for (int idx = tid; idx < 64 * 128; idx += 256) {
    int m = idx >> 7, d = idx & 127;
    size_t pix = (size_t)b * HW + p0 + m;
    Yz[idx] = yg[pix * DIN + d] * zs[pix * DIN + d];
    Ws[idx] = (_Float16)wop[idx];
  }
  __syncthreads();
  if (tid < 64) {
    float* row = &Yz[tid * 128];
    float s = 0.f;
    for (int d = 0; d < 128; ++d) s += row[d] * row[d];
    float scale = rsqrtf(s * (1.f / 128.f) + EPS);
    _Float16* hrow = (_Float16*)row;              // in-place f32 -> f16 (safe order)
    for (int d = 0; d < 128; ++d) {
      float v = row[d];
      hrow[d] = (_Float16)(v * scale * rmsw[d]);
    }
  }
  __syncthreads();
  const _Float16* Xh = (const _Float16*)Yz;       // lda = 256 halves (512B rows)
  for (int t = wid; t < 16; t += 8) {
    int tm = t >> 2, tn = t & 3;
    v8f acc = {};
#pragma unroll
    for (int kk = 0; kk < 128; kk += 32)
      acc = wmma_ab_nt(acc, &Xh[tm * 16 * 256 + kk], 256, &Ws[tn * 16 * 128 + kk], 128);
    int n = tn * 16 + r;
#pragma unroll
    for (int e = 0; e < 8; ++e) {
      int m = tm * 16 + e + 8 * hi;
      size_t pix = (size_t)b * HW + p0 + m;
      out[((size_t)b * 64 + n) * HW + p0 + m] = ctx[pix * 64 + n] + acc[e];
    }
  }
}

// ---------------------------------------------------------------------------
extern "C" void kernel_launch(void* const* d_in, const int* in_sizes, int n_in,
                              void* d_out, int out_size, void* d_ws, size_t ws_size,
                              hipStream_t stream) {
  using namespace cfg;
  const float* xx        = (const float*)d_in[0];
  const int*   rl        = (const int*)d_in[1];
  const float* score_w1  = (const float*)d_in[2];
  const float* score_b1  = (const float*)d_in[3];
  const float* score_w2  = (const float*)d_in[4];
  const float* score_b2  = (const float*)d_in[5];
  // d_in[6..9] = q_w,q_b,k_w,k_b : dead (softmax over singleton axis == 1)
  const float* v_w       = (const float*)d_in[10];
  const float* v_b       = (const float*)d_in[11];
  const float* ln_w      = (const float*)d_in[12];
  const float* ln_b      = (const float*)d_in[13];
  const float* in_proj_w = (const float*)d_in[14];
  const float* in_proj_b = (const float*)d_in[15];
  const float* conv_w    = (const float*)d_in[16];
  const float* conv_b    = (const float*)d_in[17];
  const float* dt_bias   = (const float*)d_in[18];
  const float* A_log     = (const float*)d_in[19];
  const float* ssd_D     = (const float*)d_in[20];
  const float* rms_w     = (const float*)d_in[21];
  const float* out_proj_w= (const float*)d_in[22];
  float* out = (float*)d_out;

  const int NREC = in_sizes[0] / (C * HW);   // 14
  const int Bsz  = in_sizes[1];              // 4
  const size_t MP = (size_t)Bsz * HW;        // total pixels

  char* wsp = (char*)d_ws;
  auto take = [&](size_t bytes) {
    char* p = wsp;
    wsp += (bytes + 255) & ~(size_t)255;
    return p;
  };
  float*    scores = (float*)take((size_t)NREC * HW * 4);
  float*    seqT   = (float*)take(MP * 64 * 4);
  _Float16* u_h    = (_Float16*)take(MP * 64 * 2);
  float*    ctx    = (float*)take(MP * 64 * 4);
  float*    zs     = (float*)take(MP * DIN * 4);
  float*    xBCt   = (float*)take(MP * CONVD * 4);
  float*    dtv    = (float*)take(MP * NH * 4);
  float*    dA     = (float*)take(MP * NH * 4);
  float*    xh     = (float*)take(MP * DIN * 4);
  _Float16* xdt    = (_Float16*)take(MP * DIN * 2);
  _Float16* Bh     = (_Float16*)take(MP * DST * 2);
  _Float16* Ch     = (_Float16*)take(MP * DST * 2);
  float*    Acum   = (float*)take(MP * NH * 4);
  float*    Aend   = (float*)take((size_t)Bsz * NCH * NH * 4);
  float*    states = (float*)take((size_t)Bsz * NCH * NH * 4096 * 4);
  float*    y      = (float*)take(MP * DIN * 4);

  dim3 blk(256);
  k_scores<<<dim3(HW / 64, NREC), blk, 0, stream>>>(xx, score_w1, score_b1,
                                                    score_w2, score_b2, scores);
  k_select<<<dim3(HW / 256, Bsz), blk, 0, stream>>>(xx, rl, scores, seqT);
  k_ctx_ln<<<dim3(HW / 64, Bsz), blk, 0, stream>>>(seqT, v_w, v_b, ln_w, ln_b, ctx, u_h);
  k_inproj<<<dim3(HW / 64, Bsz), blk, 0, stream>>>(u_h, in_proj_w, in_proj_b, dt_bias,
                                                   A_log, zs, xBCt, dtv, dA);
  k_conv<<<dim3(HW / 256, CONVD, Bsz), blk, 0, stream>>>(xBCt, conv_w, conv_b, dtv,
                                                         xh, xdt, Bh, Ch);
  const size_t sm_chunk = (size_t)(3 * 16384 + 65536) * sizeof(_Float16) + 256 * sizeof(float);
  k_chunk<<<dim3(NCH, NH, Bsz), blk, sm_chunk, stream>>>(dA, Bh, Ch, xdt, xh, ssd_D,
                                                         Acum, Aend, states, y);
  k_scan<<<dim3(Bsz * NH * 4096 / 256), blk, 0, stream>>>(states, Aend);
  k_yoff<<<dim3(NCH, NH, Bsz), blk, 0, stream>>>(Ch, states, Acum, y);
  const size_t sm_out = 64 * 128 * 4 + 64 * 128 * 2;
  k_out<<<dim3(HW / 64, Bsz), blk, sm_out, stream>>>(y, zs, rms_w, out_proj_w, ctx, out);
}
